// MultiHeadAttention_41532333752722
// MI455X (gfx1250) — compile-verified, matrix operand order checked by reading, not run on hardware
//
#include <hip/hip_runtime.h>

// ---------------- problem constants ----------------
constexpr int B = 2;
constexpr int S = 2048;
constexpr int E = 768;
constexpr int H = 12;
constexpr int D = 64;          // head dim
constexpr int M_TOT = B * S;   // 4096 rows for projections

typedef __attribute__((ext_vector_type(16))) __bf16 v16bf;
typedef __attribute__((ext_vector_type(8)))  float  v8f;

// -------- feature detection: Tensor Data Mover ----
#if defined(__has_builtin)
#if __has_builtin(__builtin_amdgcn_tensor_load_to_lds) && \
    __has_builtin(__builtin_amdgcn_s_wait_tensorcnt)
#define HAVE_TDM 1
#endif
#endif
#ifndef HAVE_TDM
#define HAVE_TDM 0
#endif

// ---------------- helpers ----------------

__device__ __forceinline__ void lds_fence_wave() {
  // CDNA5 split counter: wait for all outstanding LDS ops of this wave.
  asm volatile("s_wait_dscnt 0" ::: "memory");
}

#if HAVE_TDM
typedef unsigned int u32x4 __attribute__((ext_vector_type(4)));
typedef int          i32x8 __attribute__((ext_vector_type(8)));
typedef int          i32x4 __attribute__((ext_vector_type(4)));

// Issue a TDM load of a 2D bf16 tile [tile_rows x tile_cols] (row stride
// row_stride_elems in global) into LDS at lds_byte_off, padding each row by
// (pad_amount_code+1) DWORDs after every 2^(pad_interval_code+1) DWORDs.
// D# layout per CDNA5 ISA.
__device__ __forceinline__ void tdm_load_2d(const void* gptr,
                                            unsigned lds_byte_off,
                                            unsigned tile_cols,
                                            unsigned tile_rows,
                                            unsigned row_stride_elems,
                                            unsigned pad_interval_code,
                                            unsigned pad_amount_code) {
  unsigned long long ga = (unsigned long long)(uintptr_t)gptr;
  u32x4 g0;
  g0[0] = 1u;                                    // count=1, user mode, no gather
  g0[1] = lds_byte_off;                          // lds_addr
  g0[2] = (unsigned)(ga & 0xffffffffu);          // global_addr[31:0]
  g0[3] = (unsigned)((ga >> 32) & 0x01ffffffu)   // global_addr[56:32]
          | (2u << 30);                          // type=2 ("image")
  i32x8 g1;
  g1[0] = (int)((1u << 16)                       // data_size: 2 bytes
                | (1u << 20)                     // pad_enable
                | (pad_interval_code << 22) | (pad_amount_code << 25));
  g1[1] = (int)((tile_cols & 0xffffu) << 16);    // tensor_dim0[15:0] @63:48
  g1[2] = (int)(((tile_cols >> 16) & 0xffffu)    // tensor_dim0[31:16]
                | ((tile_rows & 0xffffu) << 16));// tensor_dim1[15:0] @95:80
  g1[3] = (int)(((tile_rows >> 16) & 0xffffu)    // tensor_dim1[31:16]
                | ((tile_cols & 0xffffu) << 16));// tile_dim0 @127:112
  g1[4] = (int)(tile_rows & 0xffffu);            // tile_dim1 (tile_dim2=0 -> 2D)
  g1[5] = (int)row_stride_elems;                 // tensor_dim0_stride[31:0]
  g1[6] = 0;                                     // stride hi, dim1_stride lo
  g1[7] = 0;
  i32x4 z4 = i32x4{0, 0, 0, 0};                  // groups 2/3 unused (2D)
  i32x8 z8 = i32x8{0, 0, 0, 0, 0, 0, 0, 0};
  __builtin_amdgcn_tensor_load_to_lds(g0, g1, z4, z4, z8, 0);
}

__device__ __forceinline__ unsigned lds_off_of(const void* p) {
  return (unsigned)(uintptr_t)p;  // low 32 bits of a generic LDS ptr = LDS addr
}
#endif

// Load a 16x32 bf16 fragment (A layout, or B layout from an [N][K]-stored
// tile). base points at element [0][0]; ld = row stride in elements.
// Per ISA 7.12.2: lane L holds row L&15; k-chunks at g*8 and 16+g*8 (g=L>>4).
__device__ __forceinline__ v16bf load_frag16(const __bf16* base, int ld, int lane) {
  const int row = lane & 15;
  const int g   = (lane >> 4) & 1;
  const __bf16* r = base + (size_t)row * ld;
  union { v16bf v; uint4 q[2]; } u;
  u.q[0] = *(const uint4*)(r + g * 8);
  u.q[1] = *(const uint4*)(r + 16 + g * 8);
  return u.v;
}

__device__ __forceinline__ v8f wmma_bf16(v16bf a, v16bf b, v8f c) {
  return __builtin_amdgcn_wmma_f32_16x16x32_bf16(
      /*neg_a=*/false, a, /*neg_b=*/false, b,
      /*c_mod=*/(short)0, c, /*reuse_a=*/false, /*reuse_b=*/false);
}

// ---------------- pack kernels (fp32 -> bf16, layout transforms) ----------------

__global__ void pack_x_kernel(const float* __restrict__ src,
                              __bf16* __restrict__ dst, int n) {
  int i = blockIdx.x * blockDim.x + threadIdx.x;
  if (i < n) dst[i] = (__bf16)src[i];
}

// Wt[h][sel*64+d][e] = w_sel[h][e][d]   (sel: 0=q,1=k,2=v) -> [N][K] layout
__global__ void pack_wqkv_kernel(const float* __restrict__ wq,
                                 const float* __restrict__ wk,
                                 const float* __restrict__ wv,
                                 __bf16* __restrict__ Wt) {
  int i = blockIdx.x * blockDim.x + threadIdx.x;
  int total = H * 3 * D * E;
  if (i >= total) return;
  int e  = i % E;
  int nd = (i / E) % (3 * D);
  int h  = i / (E * 3 * D);
  int sel = nd >> 6;
  int d   = nd & 63;
  const float* w = (sel == 0) ? wq : (sel == 1) ? wk : wv;
  Wt[i] = (__bf16)w[((size_t)h * E + e) * D + d];
}

// wot[n][k] = wo[k][n]  -> [N][K] layout for the output projection
__global__ void pack_wo_kernel(const float* __restrict__ wo,
                               __bf16* __restrict__ wot) {
  int i = blockIdx.x * blockDim.x + threadIdx.x;
  if (i >= E * E) return;
  int n = i / E;
  int k = i % E;
  wot[i] = (__bf16)wo[(size_t)k * E + n];
}

// ---------------- QKV projection GEMM ----------------
// Per workgroup: one (b, h, 64-row s-tile). Output = x_tile[64xE] * Wt_h^T
// giving [64 x 192] (Q|K|V concatenated). 8 waves x 6 tiles of 16x16.
// TDM staging is double-buffered: the DMA for chunk k+1 overlaps WMMA on k.
__global__ void __launch_bounds__(256)
qkv_kernel(const __bf16* __restrict__ xbf, const __bf16* __restrict__ Wt,
           const float* __restrict__ bq, const float* __restrict__ bk,
           const float* __restrict__ bv,
           __bf16* __restrict__ Q, __bf16* __restrict__ K,
           __bf16* __restrict__ V) {
  constexpr int ST = S / 64;
  constexpr int NK = E / 32;
  const int stile = blockIdx.x % ST;
  const int h     = (blockIdx.x / ST) % H;
  const int b     = blockIdx.x / (ST * H);
  const int s0    = stile * 64;

  __shared__ __align__(16) __bf16 xs[2][64][40];   // 64 rows x 32 k (+pad)
  __shared__ __align__(16) __bf16 ws[2][192][40];  // 192 n-rows x 32 k (+pad)

  const int tid  = threadIdx.x;
  const int wave = tid >> 5;
  const int lane = tid & 31;
#if HAVE_TDM
  const bool is_wave0 = ((__builtin_amdgcn_readfirstlane(tid) >> 5) == 0);
#endif

  v8f acc[6];
#pragma unroll
  for (int i = 0; i < 6; ++i) acc[i] = v8f{};

  const __bf16* xrow = xbf + ((size_t)b * S + s0) * E;
  const __bf16* wrow = Wt + (size_t)h * (3 * D) * E;

#if HAVE_TDM
  if (is_wave0) {  // prologue: prefetch chunk 0
    tdm_load_2d(xrow, lds_off_of(&xs[0][0][0]), 32, 64, E, 3, 3);
    tdm_load_2d(wrow, lds_off_of(&ws[0][0][0]), 32, 192, E, 3, 3);
  }
#endif

  for (int kc = 0; kc < NK; ++kc) {
    const int cur = kc & 1;
    const int k0  = kc * 32;
    __syncthreads();  // all reads of buf[cur^1] (iter kc-1) finished
#if HAVE_TDM
    if (is_wave0) {
      if (kc + 1 < NK) {
        tdm_load_2d(xrow + k0 + 32, lds_off_of(&xs[cur ^ 1][0][0]),
                    32, 64, E, 3, 3);
        tdm_load_2d(wrow + k0 + 32, lds_off_of(&ws[cur ^ 1][0][0]),
                    32, 192, E, 3, 3);
        __builtin_amdgcn_s_wait_tensorcnt(2);  // chunk kc done; kc+1 in flight
      } else {
        __builtin_amdgcn_s_wait_tensorcnt(0);
      }
    }
#else
    {  // stage x tile: 64x32 = 256 16B chunks, 1 per thread
      int r  = tid >> 2;
      int cc = (tid & 3) * 8;
      *(uint4*)&xs[cur][r][cc] = *(const uint4*)&xrow[(size_t)r * E + k0 + cc];
    }
#pragma unroll
    for (int i = 0; i < 3; ++i) {  // stage W tile: 192x32 = 768 chunks
      int idx = tid + i * 256;
      int r   = idx >> 2;
      int cc  = (idx & 3) * 8;
      *(uint4*)&ws[cur][r][cc] = *(const uint4*)&wrow[(size_t)r * E + k0 + cc];
    }
#endif
    __syncthreads();

#pragma unroll
    for (int i = 0; i < 6; ++i) {
      int t  = wave * 6 + i;
      int mb = t / 12, nb = t % 12;
      v16bf a  = load_frag16(&xs[cur][mb * 16][0], 40, lane);
      v16bf bf = load_frag16(&ws[cur][nb * 16][0], 40, lane);
      acc[i] = wmma_bf16(a, bf, acc[i]);
    }
  }

  // epilogue: add bias, convert to bf16, scatter to Q/K/V [B,H,S,D]
  const int g = (lane >> 4) & 1;
#pragma unroll
  for (int i = 0; i < 6; ++i) {
    int t  = wave * 6 + i;
    int mb = t / 12, nb = t % 12;
    int n   = nb * 16 + (lane & 15);
    int sel = n >> 6;
    int d   = n & 63;
    const float* bias_p = (sel == 0) ? bq : (sel == 1) ? bk : bv;
    float bias = bias_p[h * D + d];
    __bf16* dst = (sel == 0) ? Q : (sel == 1) ? K : V;
    size_t base = (((size_t)b * H + h) * S + (s0 + mb * 16 + g * 8)) * D + d;
#pragma unroll
    for (int r = 0; r < 8; ++r)
      dst[base + (size_t)r * D] = (__bf16)(acc[i][r] + bias);
  }
}

// ---------------- flash attention ----------------
// Per workgroup: one (b, h, 128-row q tile); 8 waves, each owns 16 q rows.
// kv processed in chunks of 32 with online softmax. K is TDM-double-buffered;
// V (needs transpose) is software-pipelined through a per-thread register so
// its global load overlaps compute.
__global__ void __launch_bounds__(256)
attn_kernel(const __bf16* __restrict__ Q, const __bf16* __restrict__ K,
            const __bf16* __restrict__ V, __bf16* __restrict__ Ocat) {
  constexpr int QT = S / 128;
  constexpr int NC = S / 32;
  const int qtile = blockIdx.x % QT;
  const int h     = (blockIdx.x / QT) % H;
  const int b     = blockIdx.x / (QT * H);

  const int tid  = threadIdx.x;
  const int wave = tid >> 5;
  const int lane = tid & 31;
  const int g    = (lane >> 4) & 1;
#if HAVE_TDM
  const bool is_wave0 = ((__builtin_amdgcn_readfirstlane(tid) >> 5) == 0);
#endif

  __shared__ __align__(16) __bf16 ks[2][32][72];   // K chunk, [kv][d] == [N][K]
  __shared__ __align__(16) __bf16 vts[2][64][40];  // V chunk transposed, [d][kv]
  __shared__ __align__(16) __bf16 ps[8][16][40];   // per-wave P tile 16x32

  const __bf16* Qb = Q + (((size_t)b * H + h) * S) * D;
  const __bf16* Kb = K + (((size_t)b * H + h) * S) * D;
  const __bf16* Vb = V + (((size_t)b * H + h) * S) * D;

  const int qrow0 = qtile * 128 + wave * 16;
  v16bf qa0 = load_frag16(Qb + (size_t)qrow0 * D + 0, D, lane);
  v16bf qa1 = load_frag16(Qb + (size_t)qrow0 * D + 32, D, lane);

  v8f o[4];
#pragma unroll
  for (int i = 0; i < 4; ++i) o[i] = v8f{};
  float mrow[8], lrow[8];
#pragma unroll
  for (int r = 0; r < 8; ++r) { mrow[r] = -1e30f; lrow[r] = 0.f; }

  const float scale = 0.03608439182435161f;  // 1/sqrt(E)  (matches reference)
  const float L2E   = 1.4426950408889634f;

  // V pipeline register: each thread owns one 16B chunk of the V chunk.
  const int vr = tid >> 3;        // kv row 0..31
  const int vc = (tid & 7) * 8;   // d col 0,8,...,56
  uint4 vreg = *(const uint4*)&Vb[(size_t)vr * D + vc];  // V chunk 0
#if HAVE_TDM
  if (is_wave0)  // prologue: prefetch K chunk 0
    tdm_load_2d(Kb, lds_off_of(&ks[0][0][0]), 64, 32, D, 4, 3);
#endif

  for (int kc = 0; kc < NC; ++kc) {
    const int cur = kc & 1;
    const int k0  = kc * 32;
    __syncthreads();  // all reads of buffers being overwritten are done
    {  // commit V(kc) register -> LDS transposed
      union { uint4 q; __bf16 hh[8]; } u;
      u.q = vreg;
#pragma unroll
      for (int j = 0; j < 8; ++j) vts[cur][vc + j][vr] = u.hh[j];
    }
#if HAVE_TDM
    if (is_wave0) {
      if (kc + 1 < NC) {
        tdm_load_2d(Kb + (size_t)(k0 + 32) * D, lds_off_of(&ks[cur ^ 1][0][0]),
                    64, 32, D, 4, 3);
        __builtin_amdgcn_s_wait_tensorcnt(1);  // K(kc) done; K(kc+1) in flight
      } else {
        __builtin_amdgcn_s_wait_tensorcnt(0);
      }
    }
#else
    {  // stage K chunk synchronously
      *(uint4*)&ks[cur][vr][vc] = *(const uint4*)&Kb[(size_t)(k0 + vr) * D + vc];
    }
#endif
    __syncthreads();

    if (kc + 1 < NC)  // prefetch next V chunk; overlaps the compute below
      vreg = *(const uint4*)&Vb[(size_t)(k0 + 32 + vr) * D + vc];

    // scores: 16 q rows x 32 kv cols
    v8f sc0 = v8f{}, sc1 = v8f{};
    sc0 = wmma_bf16(qa0, load_frag16(&ks[cur][0][0], 72, lane), sc0);
    sc0 = wmma_bf16(qa1, load_frag16(&ks[cur][0][32], 72, lane), sc0);
    sc1 = wmma_bf16(qa0, load_frag16(&ks[cur][16][0], 72, lane), sc1);
    sc1 = wmma_bf16(qa1, load_frag16(&ks[cur][16][32], 72, lane), sc1);

    // online softmax over the 32 columns
#pragma unroll
    for (int r = 0; r < 8; ++r) {
      float s0 = sc0[r] * scale;
      float s1 = sc1[r] * scale;
      float rm = fmaxf(s0, s1);
#pragma unroll
      for (int off = 8; off >= 1; off >>= 1)
        rm = fmaxf(rm, __shfl_xor(rm, off, 32));
      float mnew  = fmaxf(mrow[r], rm);
      float alpha = exp2f((mrow[r] - mnew) * L2E);
      float p0    = exp2f((s0 - mnew) * L2E);
      float p1    = exp2f((s1 - mnew) * L2E);
      float rs    = p0 + p1;
#pragma unroll
      for (int off = 8; off >= 1; off >>= 1)
        rs += __shfl_xor(rs, off, 32);
      lrow[r] = lrow[r] * alpha + rs;
      mrow[r] = mnew;
#pragma unroll
      for (int nb = 0; nb < 4; ++nb) o[nb][r] *= alpha;
      // C layout -> LDS (row M = r+8g, cols lane&15 / 16+(lane&15))
      ps[wave][r + 8 * g][lane & 15]        = (__bf16)p0;
      ps[wave][r + 8 * g][16 + (lane & 15)] = (__bf16)p1;
    }
    lds_fence_wave();  // same-wave cross-lane LDS visibility

    v16bf pa = load_frag16(&ps[wave][0][0], 40, lane);
#pragma unroll
    for (int nb = 0; nb < 4; ++nb)
      o[nb] = wmma_bf16(pa, load_frag16(&vts[cur][nb * 16][0], 40, lane), o[nb]);
  }

  // epilogue: normalize and write concat-head output [B,S,E] bf16
  __bf16* Ob = Ocat + ((size_t)b * S) * E + (size_t)h * D;
#pragma unroll
  for (int nb = 0; nb < 4; ++nb) {
    int d = nb * 16 + (lane & 15);
#pragma unroll
    for (int r = 0; r < 8; ++r) {
      int srow = qrow0 + r + 8 * g;
      Ob[(size_t)srow * E + d] = (__bf16)(o[nb][r] / lrow[r]);
    }
  }
}

// ---------------- output projection GEMM ----------------
// out[M_TOT x E] = Ocat[M_TOT x E] * wo[E x E] + bo  (fp32 out)
__global__ void __launch_bounds__(256)
proj_kernel(const __bf16* __restrict__ Ocat, const __bf16* __restrict__ wot,
            const float* __restrict__ bo, float* __restrict__ out) {
  constexpr int NT = E / 128;  // 6
  constexpr int NK = E / 32;
  const int nt = blockIdx.x % NT;
  const int mt = blockIdx.x / NT;
  const int m0 = mt * 64, n0 = nt * 128;

  __shared__ __align__(16) __bf16 as[2][64][40];
  __shared__ __align__(16) __bf16 bs[2][128][40];

  const int tid  = threadIdx.x;
  const int wave = tid >> 5;
  const int lane = tid & 31;
#if HAVE_TDM
  const bool is_wave0 = ((__builtin_amdgcn_readfirstlane(tid) >> 5) == 0);
#endif

  const __bf16* arow = Ocat + (size_t)m0 * E;
  const __bf16* brow = wot + (size_t)n0 * E;

  v8f acc[4];
#pragma unroll
  for (int i = 0; i < 4; ++i) acc[i] = v8f{};

#if HAVE_TDM
  if (is_wave0) {  // prologue: prefetch chunk 0
    tdm_load_2d(arow, lds_off_of(&as[0][0][0]), 32, 64, E, 3, 3);
    tdm_load_2d(brow, lds_off_of(&bs[0][0][0]), 32, 128, E, 3, 3);
  }
#endif

  for (int kc = 0; kc < NK; ++kc) {
    const int cur = kc & 1;
    const int k0  = kc * 32;
    __syncthreads();
#if HAVE_TDM
    if (is_wave0) {
      if (kc + 1 < NK) {
        tdm_load_2d(arow + k0 + 32, lds_off_of(&as[cur ^ 1][0][0]),
                    32, 64, E, 3, 3);
        tdm_load_2d(brow + k0 + 32, lds_off_of(&bs[cur ^ 1][0][0]),
                    32, 128, E, 3, 3);
        __builtin_amdgcn_s_wait_tensorcnt(2);
      } else {
        __builtin_amdgcn_s_wait_tensorcnt(0);
      }
    }
#else
    {
      int r  = tid >> 2;
      int cc = (tid & 3) * 8;
      *(uint4*)&as[cur][r][cc] = *(const uint4*)&arow[(size_t)r * E + k0 + cc];
    }
#pragma unroll
    for (int i = 0; i < 2; ++i) {
      int idx = tid + i * 256;
      int r   = idx >> 2;
      int cc  = (idx & 3) * 8;
      *(uint4*)&bs[cur][r][cc] = *(const uint4*)&brow[(size_t)r * E + k0 + cc];
    }
#endif
    __syncthreads();

#pragma unroll
    for (int i = 0; i < 4; ++i) {
      int t  = wave * 4 + i;
      int mb = t / 8, nb = t % 8;
      v16bf a  = load_frag16(&as[cur][mb * 16][0], 40, lane);
      v16bf bf = load_frag16(&bs[cur][nb * 16][0], 40, lane);
      acc[i] = wmma_bf16(a, bf, acc[i]);
    }
  }

  const int g = (lane >> 4) & 1;
#pragma unroll
  for (int i = 0; i < 4; ++i) {
    int t  = wave * 4 + i;
    int mb = t / 8, nb = t % 8;
    int coln = n0 + nb * 16 + (lane & 15);
    float bias = bo[coln];
#pragma unroll
    for (int r = 0; r < 8; ++r) {
      int row = m0 + mb * 16 + r + 8 * g;
      out[(size_t)row * E + coln] = acc[i][r] + bias;
    }
  }
}

// ---------------- launch ----------------
extern "C" void kernel_launch(void* const* d_in, const int* in_sizes, int n_in,
                              void* d_out, int out_size, void* d_ws,
                              size_t ws_size, hipStream_t stream) {
  const float* x  = (const float*)d_in[0];
  const float* wq = (const float*)d_in[1];
  const float* bq = (const float*)d_in[2];
  const float* wk = (const float*)d_in[3];
  const float* bk = (const float*)d_in[4];
  const float* wv = (const float*)d_in[5];
  const float* bv = (const float*)d_in[6];
  const float* wo = (const float*)d_in[7];
  const float* bo = (const float*)d_in[8];
  float* out = (float*)d_out;

  // workspace carve-up (256B aligned)
  size_t off = 0;
  auto take = [&](size_t bytes) -> void* {
    void* p = (char*)d_ws + off;
    off += (bytes + 255) & ~(size_t)255;
    return p;
  };
  __bf16* xbf  = (__bf16*)take((size_t)B * S * E * 2);
  __bf16* Wt   = (__bf16*)take((size_t)H * 3 * D * E * 2);
  __bf16* wot  = (__bf16*)take((size_t)E * E * 2);
  __bf16* Qb   = (__bf16*)take((size_t)B * H * S * D * 2);
  __bf16* Kb   = (__bf16*)take((size_t)B * H * S * D * 2);
  __bf16* Vb   = (__bf16*)take((size_t)B * H * S * D * 2);
  __bf16* Ocat = (__bf16*)take((size_t)B * S * E * 2);
  (void)ws_size;

  {
    int n = B * S * E;
    pack_x_kernel<<<(n + 255) / 256, 256, 0, stream>>>(x, xbf, n);
  }
  {
    int n = H * 3 * D * E;
    pack_wqkv_kernel<<<(n + 255) / 256, 256, 0, stream>>>(wq, wk, wv, Wt);
  }
  {
    int n = E * E;
    pack_wo_kernel<<<(n + 255) / 256, 256, 0, stream>>>(wo, wot);
  }

  qkv_kernel<<<B * H * (S / 64), 256, 0, stream>>>(xbf, Wt, bq, bk, bv,
                                                   Qb, Kb, Vb);
  attn_kernel<<<B * H * (S / 128), 256, 0, stream>>>(Qb, Kb, Vb, Ocat);
  proj_kernel<<<(M_TOT / 64) * (E / 128), 256, 0, stream>>>(Ocat, wot, bo, out);
}
